// Model_31894427140232
// MI455X (gfx1250) — matrix operand order repaired and checked by reading, not compiled, not run on hardware
//
#include <hip/hip_runtime.h>
#include <hip/hip_bf16.h>
#include <math.h>

typedef __attribute__((ext_vector_type(16))) __bf16 v16bf;
typedef __attribute__((ext_vector_type(4)))  __bf16 v4bf;
typedef __attribute__((ext_vector_type(2)))  __bf16 v2bf;
typedef __attribute__((ext_vector_type(8)))  float  v8f;

#define B_    128
#define SEQ_  200
#define NROW  (B_ * SEQ_)   // 25600

__device__ __forceinline__ float relu6f(float x) { return fminf(fmaxf(x, 0.f), 6.f); }

// ---------------------------------------------------------------------------
// Weight pre-pack: W (N,K) fp32 row-major -> Wp bf16 in WMMA-B fragment order:
//   Wp[((kb*(N/16) + n16)*32 + lane)*16 + e]   (kb = k/32 block)
// B-matrix layout: lane <-> N (lane&15), h = lane>>4; e<8 -> k=h*8+e ; e>=8 -> k=16+h*8+(e-8).
// One dword (2 bf16, consecutive k) per thread.
// ---------------------------------------------------------------------------
__global__ __launch_bounds__(256) void wpack_kernel(
    const float* __restrict__ W, __bf16* __restrict__ Wp, int N, int K)
{
  int idx = blockIdx.x * 256 + threadIdx.x;     // pair index, N*K/2 total
  int total = (N * K) >> 1;
  if (idx >= total) return;
  int ep  = idx & 7;                // element pair within 16
  int kk  = (idx >> 3) & 31;        // lane (n = kk&15, h = kk>>4)
  int t2  = idx >> 8;
  int nb  = N >> 4;
  int n16 = t2 % nb;
  int kb  = t2 / nb;
  int k   = (kb << 5) + ((ep >> 2) << 4) + ((kk >> 4) << 3) + ((ep & 3) << 1);
  int n0  = (n16 << 4) + (kk & 15);
  v2bf p = { (__bf16)W[(size_t)n0 * K + k], (__bf16)W[(size_t)n0 * K + k + 1] };
  *(v2bf*)&Wp[(size_t)idx * 2] = p;
}

// ---------------------------------------------------------------------------
// WMMA GEMM: C[M,N] = A[M,K] @ W[N,K]^T + bias[N]
// A fp32 (staged fp32->bf16 through LDS in fragment-major order),
// W pre-packed bf16 fragments read straight from global (L2-resident).
// 256 threads = 8 waves; block tile 128x64; per-wave 32x32 (2x2 16x16 frags).
// M%128==0, N%64==0, K%32==0.
// ---------------------------------------------------------------------------
__global__ __launch_bounds__(256) void gemm_bias_wmma(
    const float* __restrict__ A, const __bf16* __restrict__ Wp,
    const float* __restrict__ bias, float* __restrict__ C,
    int M, int N, int K)
{
  constexpr int TM = 128, TK = 32;
  // fragment-major A tile: [fr(8)][lane(32)][e(16)]  -> 8 KB
  __shared__ alignas(32) __bf16 As[8 * 32 * 16];

  const int tid  = threadIdx.x;
  const int lane = tid & 31;
  const int wave = tid >> 5;
  const int wm   = (wave >> 1) * 32;      // wave row offset (4 wave-rows)
  const int wn   = (wave & 1) * 32;       // wave col offset (2 wave-cols)
  const int bm   = blockIdx.x * TM;
  const int bn   = blockIdx.y * 64;
  const int half = lane >> 4;
  const int r16  = lane & 15;
  const int nb16 = N >> 4;
  const int frA0 = wm >> 4;               // this wave's A fragment blocks

  v8f acc[2][2] = {};

  for (int kb = 0; kb < (K >> 5); ++kb) {
    const int k0 = kb << 5;
    __syncthreads();
    // ---- stage A tile (128x32 fp32) into fragment-major bf16 LDS ---------
    // A-matrix 16x32 bf16 layout: lane(r16,h): e<8 -> k=h*8+e ; e>=8 -> k=16+h*8+(e-8)
#pragma unroll
    for (int i = 0; i < 4; ++i) {
      int idx = i * 256 + tid;            // 1024 float4 chunks
      int r  = idx >> 3;                  // tile row 0..127
      int c4 = (idx & 7) * 4;             // k offset 0..28 (stays in one octet)
      float4 v = *(const float4*)(A + (size_t)(bm + r) * K + (k0 + c4));
      int fr  = r >> 4, rr = r & 15;
      int oct = c4 >> 3;
      int ln  = rr + ((oct & 1) << 4);
      int e0  = ((oct >> 1) << 3) + (c4 & 7);
      v4bf p = { (__bf16)v.x, (__bf16)v.y, (__bf16)v.z, (__bf16)v.w };
      *(v4bf*)&As[(fr * 32 + ln) * 16 + e0] = p;   // one ds_store_b64
    }
    if (kb + 1 < (K >> 5))
      __builtin_prefetch(A + (size_t)(bm + (tid >> 1)) * K + (k0 + TK), 0, 1);
    __syncthreads();

    // ---- A fragments: one aligned 32B vector load per frag ---------------
    v16bf af0 = *(const v16bf*)&As[((frA0    ) * 32 + lane) * 16];
    v16bf af1 = *(const v16bf*)&As[((frA0 + 1) * 32 + lane) * 16];
    // ---- B fragments: coalesced 32B loads from packed weights ------------
    const __bf16* wp =
        Wp + (((size_t)kb * nb16 + ((bn + wn) >> 4)) * 512) + lane * 16;
    v16bf bf0 = *(const v16bf*)(wp);
    v16bf bf1 = *(const v16bf*)(wp + 512);

#if defined(__HIP_DEVICE_COMPILE__)
    acc[0][0] = __builtin_amdgcn_wmma_f32_16x16x32_bf16(
        false, af0, false, bf0, (short)0, acc[0][0], false, false);
    acc[0][1] = __builtin_amdgcn_wmma_f32_16x16x32_bf16(
        false, af0, false, bf1, (short)0, acc[0][1], false, false);
    acc[1][0] = __builtin_amdgcn_wmma_f32_16x16x32_bf16(
        false, af1, false, bf0, (short)0, acc[1][0], false, false);
    acc[1][1] = __builtin_amdgcn_wmma_f32_16x16x32_bf16(
        false, af1, false, bf1, (short)0, acc[1][1], false, false);
#endif
  }

  // D layout: VGPR e, lanes 0-15 -> M=e, N=lane; lanes 16-31 -> M=e+8
#pragma unroll
  for (int fm = 0; fm < 2; ++fm)
#pragma unroll
    for (int fn = 0; fn < 2; ++fn) {
      int col = bn + wn + fn * 16 + r16;
      float bv = bias[col];
#pragma unroll
      for (int e = 0; e < 8; ++e) {
        int row = bm + wm + fm * 16 + half * 8 + e;
        C[(size_t)row * N + col] = acc[fm][fn][e] + bv;
      }
    }
}

// ---------------------------------------------------------------------------
// Fused residual + layernorm over D=512 (one block per row, in-place on x).
// resBcast==1: residual is a per-batch row (B,512) broadcast over SEQ.
// ---------------------------------------------------------------------------
__global__ __launch_bounds__(256) void ln_res_kernel(
    float* __restrict__ x, const float* __restrict__ res, int resBcast,
    const float* __restrict__ g, const float* __restrict__ bb)
{
  int row = blockIdx.x, t = threadIdx.x;
  __shared__ float rs[256], rq[256];
  size_t base  = (size_t)row * 512;
  size_t rbase = resBcast ? (size_t)(row / SEQ_) * 512 : base;
  float v0 = x[base + t]       + res[rbase + t];
  float v1 = x[base + t + 256] + res[rbase + t + 256];
  rs[t] = v0 + v1;
  rq[t] = v0 * v0 + v1 * v1;
  __syncthreads();
  for (int off = 128; off > 0; off >>= 1) {
    if (t < off) { rs[t] += rs[t + off]; rq[t] += rq[t + off]; }
    __syncthreads();
  }
  float mean = rs[0] * (1.f / 512.f);
  float var  = rq[0] * (1.f / 512.f) - mean * mean;
  float inv  = rsqrtf(var + 1e-5f);
  x[base + t]       = (v0 - mean) * inv * g[t]       + bb[t];
  x[base + t + 256] = (v1 - mean) * inv * g[t + 256] + bb[t + 256];
}

// ---------------------------------------------------------------------------
// Preamble: loc MLP, piece embedding gather
// ---------------------------------------------------------------------------
__global__ __launch_bounds__(512) void loc_kernel(
    const float* __restrict__ xyrot, const float* __restrict__ w1,
    const float* __restrict__ b1, const float* __restrict__ w2,
    const float* __restrict__ b2, float* __restrict__ loc)
{
  __shared__ float l16[16];
  int b = blockIdx.x, t = threadIdx.x;
  if (t < 16) {
    float s = b1[t];
    for (int j = 0; j < 3; ++j) s += xyrot[b * 3 + j] * w1[t * 3 + j];
    l16[t] = s;
  }
  __syncthreads();
  float s = b2[t];
  for (int j = 0; j < 16; ++j) s += l16[j] * w2[t * 16 + j];
  loc[(size_t)b * 512 + t] = s;
}

__global__ __launch_bounds__(512) void pemb_kernel(
    const int* __restrict__ queue, const int* __restrict__ remains,
    const float* __restrict__ piece_emb, const float* __restrict__ loc,
    float* __restrict__ pembB)
{
  int bi = blockIdx.x;          // b*14 + i
  int b = bi / 14, i = bi % 14;
  int t = threadIdx.x;
  int piece = (i < 7) ? queue[b * 7 + i] : remains[b * 7 + (i - 7)];
  float v = piece_emb[(size_t)piece * 512 + t];
  if (i == 0) v += loc[(size_t)b * 512 + t];
  pembB[(size_t)bi * 512 + t] = v;
}

// ---------------------------------------------------------------------------
// Garbage path: padded conv1d(512 ch, k=3) kept as per-(b,c) 10-vector
// ---------------------------------------------------------------------------
__global__ __launch_bounds__(512) void gconv_kernel(
    const float* __restrict__ garbage, const float* __restrict__ g1_w,
    const float* __restrict__ g1_b, float* __restrict__ gconv)
{
  int b = blockIdx.x, c = threadIdx.x;
  const float* gw = g1_w + c * 3;
  const float* gb = garbage + b * 10;
  for (int t10 = 0; t10 < 10; ++t10) {
    float s = g1_b[c];
#pragma unroll
    for (int j = 0; j < 3; ++j) {
      int idx = t10 + j - 1;
      if (idx >= 0 && idx < 10) s += gb[idx] * gw[j];
    }
    gconv[((size_t)b * 512 + c) * 10 + t10] = s;
  }
}

// ---------------------------------------------------------------------------
// CNN: dw1(3x3 valid)+pw1+bn1+relu6 -> a1 ; dw2(3x3 pad1 depthwise) -> d2 ;
// pw2+bn2+relu6 fused with pos-encoding and garbage projection -> state
// ---------------------------------------------------------------------------
__global__ __launch_bounds__(256) void cnn_a_kernel(
    const float* __restrict__ board, const float* __restrict__ dw1_w,
    const float* __restrict__ dw1_b, const float* __restrict__ pw1_w,
    const float* __restrict__ pw1_b, const float* __restrict__ bn1_g,
    const float* __restrict__ bn1_b, float* __restrict__ a1)
{
  int b = blockIdx.x, t = threadIdx.x;
  if (t >= 200) return;
  int y = t / 10, x = t % 10;
  const float* bp = board + (size_t)b * 22 * 12;
  float c1 = dw1_b[0];
#pragma unroll
  for (int dy = 0; dy < 3; ++dy)
#pragma unroll
    for (int dx = 0; dx < 3; ++dx)
      c1 += bp[(y + dy) * 12 + (x + dx)] * dw1_w[dy * 3 + dx];
  const float bninv = rsqrtf(1.0f + 1e-5f);
#pragma unroll
  for (int c = 0; c < 16; ++c) {
    float v = (c1 * pw1_w[c] + pw1_b[c]) * (bn1_g[c] * bninv) + bn1_b[c];
    a1[((size_t)b * 16 + c) * 200 + t] = relu6f(v);
  }
}

__global__ __launch_bounds__(256) void cnn_b_kernel(
    const float* __restrict__ a1, const float* __restrict__ dw2_w,
    const float* __restrict__ dw2_b, float* __restrict__ d2)
{
  int bc = blockIdx.x;
  int b = bc >> 4, c = bc & 15;
  int t = threadIdx.x;
  if (t >= 200) return;
  int y = t / 10, x = t % 10;
  const float* ap = a1 + ((size_t)b * 16 + c) * 200;
  const float* wp = dw2_w + c * 9;
  float s = dw2_b[c];
#pragma unroll
  for (int dy = -1; dy <= 1; ++dy)
#pragma unroll
    for (int dx = -1; dx <= 1; ++dx) {
      int yy = y + dy, xx = x + dx;
      if (yy >= 0 && yy < 20 && xx >= 0 && xx < 10)
        s += ap[yy * 10 + xx] * wp[(dy + 1) * 3 + (dx + 1)];
    }
  d2[((size_t)b * 16 + c) * 200 + t] = s;
}

__global__ __launch_bounds__(512) void cnn_state_kernel(
    const float* __restrict__ d2, const float* __restrict__ pw2_w,
    const float* __restrict__ pw2_b, const float* __restrict__ bn2_g,
    const float* __restrict__ bn2_b, const float* __restrict__ gconv,
    const float* __restrict__ g2_w, const float* __restrict__ g2_b,
    float* __restrict__ state)
{
  int bs = blockIdx.x;          // b*200 + s
  int b = bs / 200, s = bs % 200;
  int f = threadIdx.x;
  float acc = pw2_b[f];
  const float* dp = d2 + (size_t)b * 16 * 200 + s;
  const float* wp = pw2_w + f * 16;
#pragma unroll
  for (int c = 0; c < 16; ++c) acc += dp[c * 200] * wp[c];
  const float bninv = rsqrtf(1.0f + 1e-5f);
  float v = relu6f(acc * (bn2_g[f] * bninv) + bn2_b[f]);
  int i = f >> 1;
  float dv = expf(-(float)(2 * i) * (9.210340371976184f / 512.0f));
  float ang = (float)s * dv;
  float pe = (f & 1) ? cosf(ang) : sinf(ang);
  float gg = g2_b[s];
  const float* gc = gconv + ((size_t)b * 512 + f) * 10;
  const float* g2r = g2_w + s * 10;
#pragma unroll
  for (int t10 = 0; t10 < 10; ++t10) gg += gc[t10] * g2r[t10];
  state[(size_t)bs * 512 + f] = v + pe + gg;
}

// ---------------------------------------------------------------------------
// Loop helper: gather the broadcast v-row for piece i (contiguous reshape)
// ---------------------------------------------------------------------------
__global__ __launch_bounds__(512) void vrow_gather_kernel(
    const float* __restrict__ vAll, int i, float* __restrict__ vrow)
{
  int b = blockIdx.x, j = threadIdx.x;
  vrow[(size_t)b * 512 + j] = vAll[((size_t)(b * 14 + i)) * 512 + j];
}

// ---------------------------------------------------------------------------
// final_bias MLP: stage A (per-token 4-dot), stage B (per-batch trunk),
// stage C (w4 expansion to (B,8,1400))
// ---------------------------------------------------------------------------
__global__ __launch_bounds__(256) void bias_h0_kernel(
    const float* __restrict__ x, const float* __restrict__ w0,
    float* __restrict__ h0)
{
  int row = blockIdx.x, t = threadIdx.x;
  int c = t >> 6, j = t & 63;
  __shared__ float red[256];
  const float* xr = x + (size_t)row * 512;
  const float* wr = w0 + c * 512;
  float s = 0.f;
  for (int k = j; k < 512; k += 64) s += xr[k] * wr[k];
  red[t] = s;
  __syncthreads();
  for (int off = 32; off > 0; off >>= 1) {
    if (j < off) red[t] += red[t + off];
    __syncthreads();
  }
  if (j == 0) {
    int b = row / SEQ_, sq = row % SEQ_;
    h0[(size_t)b * 800 + sq * 4 + c] = red[t];
  }
}

__global__ __launch_bounds__(256) void bias_mid_kernel(
    const float* __restrict__ h0, const float* __restrict__ w1,
    const float* __restrict__ w2, const float* __restrict__ b2,
    const float* __restrict__ g1, const float* __restrict__ be1,
    const float* __restrict__ w3, const float* __restrict__ b3,
    const float* __restrict__ g2, const float* __restrict__ be2,
    float* __restrict__ h3out)
{
  int b = blockIdx.x, t = threadIdx.x;
  __shared__ float s1[32];
  __shared__ float s2[128];
  __shared__ float h3s[1024];
  __shared__ float mv[2];
  __shared__ float gm[8], gi[8];

  if (t < 32) {
    const float* h = h0 + (size_t)b * 800;
    const float* w = w1 + (size_t)t * 800;
    float s = 0.f;
    for (int k = 0; k < 800; ++k) s += h[k] * w[k];
    s1[t] = relu6f(s);
  }
  __syncthreads();
  if (t < 128) {
    const float* w = w2 + t * 32;
    float s = b2[t];
#pragma unroll
    for (int k = 0; k < 32; ++k) s += s1[k] * w[k];
    s2[t] = s;
  }
  __syncthreads();
  if (t == 0) {
    float m = 0.f;
    for (int k = 0; k < 128; ++k) m += s2[k];
    m *= (1.f / 128.f);
    float v = 0.f;
    for (int k = 0; k < 128; ++k) { float d = s2[k] - m; v += d * d; }
    v *= (1.f / 128.f);
    mv[0] = m; mv[1] = rsqrtf(v + 1e-5f);
  }
  __syncthreads();
  if (t < 128) s2[t] = relu6f((s2[t] - mv[0]) * mv[1] * g1[t] + be1[t]);
  __syncthreads();
#pragma unroll
  for (int ii = 0; ii < 4; ++ii) {
    int o = t + 256 * ii;
    const float* w = w3 + (size_t)o * 128;
    float s = b3[o];
    for (int j = 0; j < 128; ++j) s += s2[j] * w[j];
    h3s[o] = s;
  }
  __syncthreads();
  if (t < 8) {
    float m = 0.f;
    for (int j = 0; j < 128; ++j) m += h3s[t * 128 + j];
    m *= (1.f / 128.f);
    float v = 0.f;
    for (int j = 0; j < 128; ++j) { float d = h3s[t * 128 + j] - m; v += d * d; }
    v *= (1.f / 128.f);
    gm[t] = m; gi[t] = rsqrtf(v + 1e-5f);
  }
  __syncthreads();
#pragma unroll
  for (int ii = 0; ii < 4; ++ii) {
    int o = t + 256 * ii;
    int grp = o >> 7, j = o & 127;
    h3out[(size_t)b * 1024 + o] =
        relu6f((h3s[o] - gm[grp]) * gi[grp] * g2[j] + be2[j]);
  }
}

__global__ __launch_bounds__(256) void bias_w4_kernel(
    const float* __restrict__ h3, const float* __restrict__ w4,
    const float* __restrict__ b4, float* __restrict__ fbias)
{
  int idx = blockIdx.x * 256 + threadIdx.x;   // B*8*1400 = 1433600 exact
  int b = idx / 11200;
  int rem = idx % 11200;
  int n = rem / 1400;
  int m = rem % 1400;
  const float* h = h3 + (size_t)b * 1024 + n * 128;
  const float* w = w4 + (size_t)m * 128;
  float s = b4[m];
  for (int j = 0; j < 128; ++j) s += h[j] * w[j];
  fbias[idx] = s;
}

// ---------------------------------------------------------------------------
// Final attention (Kn=7): one block per (b,s), one wave per head (wave32).
// Overwrites the q buffer in place with the attention output.
// ---------------------------------------------------------------------------
__global__ __launch_bounds__(256) void final_attn_kernel(
    float* __restrict__ q, const float* __restrict__ kAll,
    const float* __restrict__ vAll, const float* __restrict__ fbias)
{
  int row = blockIdx.x;
  int b = row / SEQ_, s = row % SEQ_;
  int t = threadIdx.x;
  int n = t >> 5, lane = t & 31;
  float* qp = q + (size_t)row * 512 + n * 64;
  float q0 = qp[lane], q1 = qp[lane + 32];
  float sc[7];
#pragma unroll
  for (int k7 = 0; k7 < 7; ++k7) {
    const float* kp = kAll + ((size_t)(b * 14 + 7 + k7)) * 512 + n * 64;
    float p = q0 * kp[lane] + q1 * kp[lane + 32];
#pragma unroll
    for (int off = 16; off > 0; off >>= 1) p += __shfl_xor(p, off, 32);
    sc[k7] = p * 0.125f + fbias[(size_t)b * 11200 + n * 1400 + s * 7 + k7];
  }
  float mx = sc[0];
#pragma unroll
  for (int k7 = 1; k7 < 7; ++k7) mx = fmaxf(mx, sc[k7]);
  float den = 0.f;
#pragma unroll
  for (int k7 = 0; k7 < 7; ++k7) { sc[k7] = expf(sc[k7] - mx); den += sc[k7]; }
  float inv = 1.f / den;
  float o0 = 0.f, o1 = 0.f;
#pragma unroll
  for (int k7 = 0; k7 < 7; ++k7) {
    const float* vp = vAll + ((size_t)(b * 14 + 7 + k7)) * 512 + n * 64;
    float w = sc[k7] * inv;
    o0 += w * vp[lane];
    o1 += w * vp[lane + 32];
  }
  qp[lane] = o0;
  qp[lane + 32] = o1;
}

// ---------------------------------------------------------------------------
// Max pool: (B,200,512) -> reshape (B,20,5,2,256,2).max(3,5) -> (B,50,512)
// ---------------------------------------------------------------------------
__global__ __launch_bounds__(512) void pool_kernel(
    const float* __restrict__ st, float* __restrict__ pooled)
{
  int br = blockIdx.x;          // b*50 + r
  int b = br / 50, r = br % 50;
  int idx = r * 512 + threadIdx.x;
  int a  = idx / 1280;
  int rm = idx % 1280;
  int bb = rm / 256;
  int j  = rm % 256;
  int sb = a * 10 + bb * 2;
  const float* p = st + ((size_t)b * 200 + sb) * 512 + j * 2;
  float m = fmaxf(fmaxf(p[0], p[1]), fmaxf(p[512], p[513]));
  pooled[(size_t)br * 512 + threadIdx.x] = m;
}

// ---------------------------------------------------------------------------
extern "C" void kernel_launch(void* const* d_in, const int* in_sizes, int n_in,
                              void* d_out, int out_size, void* d_ws, size_t ws_size,
                              hipStream_t stream)
{
  (void)in_sizes; (void)n_in; (void)out_size; (void)ws_size;
  auto F = [&](int i) { return (const float*)d_in[i]; };
  const float* xyrot    = F(0);
  const int*   queue    = (const int*)d_in[1];
  const int*   remains  = (const int*)d_in[2];
  const float* board    = F(3);
  const float* garbage  = F(4);
  const float *loc1_w = F(5),  *loc1_b = F(6);
  const float *loc2_w = F(7),  *loc2_b = F(8);
  const float *piece_emb = F(9);
  const float *rot_w = F(10), *rot_b = F(11);
  const float *g1_w = F(12),  *g1_b = F(13);
  const float *g2_w = F(14),  *g2_b = F(15);
  const float *q_w = F(16),   *q_b = F(17);
  const float *k_w = F(18),   *k_b = F(19);
  const float *v_w = F(20),   *v_b = F(21);
  const float *qkv_w = F(22), *qkv_b = F(23);
  // attn_bias leaves (24..35) are provably dead: softmax over length-1 axis == 1
  const float *fb_w0 = F(36), *fb_w1 = F(37), *fb_w2 = F(38), *fb_b2 = F(39);
  const float *fb_g1 = F(40), *fb_be1 = F(41), *fb_w3 = F(42), *fb_b3 = F(43);
  const float *fb_g2 = F(44), *fb_be2 = F(45), *fb_w4 = F(46), *fb_b4 = F(47);
  const float *queue_w = F(48), *queue_b = F(49);
  const float *an_g = F(50), *an_b = F(51);
  const float *fn_g = F(52), *fn_b = F(53);
  const float *remain_w = F(54), *remain_b = F(55);
  const float *fan_g = F(56), *fan_b = F(57);
  const float *ffn_g = F(58), *ffn_b = F(59);
  const float *dw1_w = F(60), *dw1_b = F(61);
  const float *pw1_w = F(62), *pw1_b = F(63);
  const float *bn1_g = F(64), *bn1_b = F(65);
  const float *dw2_w = F(66), *dw2_b = F(67);
  const float *pw2_w = F(68), *pw2_b = F(69);
  const float *bn2_g = F(70), *bn2_b = F(71);
  const float *last_w = F(72), *last_b = F(73);
  float* out = (float*)d_out;

  float* ws = (float*)d_ws;
  float* state  = ws; ws += (size_t)NROW * 512;
  float* tmpA   = ws; ws += (size_t)NROW * 512;
  float* tmpB   = ws; ws += (size_t)NROW * 512;
  float* pembB  = ws; ws += (size_t)B_ * 14 * 512;
  float* rots   = ws; ws += (size_t)B_ * 14 * 2048;
  float* kAll   = ws; ws += (size_t)B_ * 56 * 128;
  float* vAll   = ws; ws += (size_t)B_ * 56 * 128;
  float* locB   = ws; ws += (size_t)B_ * 512;
  float* gconv  = ws; ws += (size_t)B_ * 512 * 10;
  float* fh0    = ws; ws += (size_t)B_ * 800;
  float* fh3    = ws; ws += (size_t)B_ * 1024;
  float* fbias  = ws; ws += (size_t)B_ * 11200;
  float* vrow   = ws; ws += (size_t)B_ * 512;
  float* nsrow  = ws; ws += (size_t)B_ * 512;
  float* a1     = ws; ws += (size_t)B_ * 16 * 200;
  float* d2b    = ws; ws += (size_t)B_ * 16 * 200;
  float* pooled = ws; ws += (size_t)B_ * 50 * 512;
  // packed bf16 weights (sizes in floats = N*K/2; all multiples of 256B)
  __bf16* rotP    = (__bf16*)ws; ws += (size_t)2048 * 512 / 2;
  __bf16* kP      = (__bf16*)ws; ws += (size_t)128 * 512 / 2;
  __bf16* vP      = (__bf16*)ws; ws += (size_t)128 * 512 / 2;
  __bf16* qkvP    = (__bf16*)ws; ws += (size_t)512 * 512 / 2;
  __bf16* queueP  = (__bf16*)ws; ws += (size_t)512 * 512 / 2;
  __bf16* qP      = (__bf16*)ws; ws += (size_t)512 * 512 / 2;
  __bf16* remainP = (__bf16*)ws; ws += (size_t)512 * 512 / 2;
  __bf16* lastP   = (__bf16*)ws; ws += (size_t)512 * 512 / 2;

  dim3 blk(256);

  // ---- pack weights into WMMA-B fragment order (one-time, L2-resident) ----
  wpack_kernel<<<2048, blk, 0, stream>>>(rot_w,    rotP,    2048, 512);
  wpack_kernel<<<128,  blk, 0, stream>>>(k_w,      kP,      128,  512);
  wpack_kernel<<<128,  blk, 0, stream>>>(v_w,      vP,      128,  512);
  wpack_kernel<<<512,  blk, 0, stream>>>(qkv_w,    qkvP,    512,  512);
  wpack_kernel<<<512,  blk, 0, stream>>>(queue_w,  queueP,  512,  512);
  wpack_kernel<<<512,  blk, 0, stream>>>(q_w,      qP,      512,  512);
  wpack_kernel<<<512,  blk, 0, stream>>>(remain_w, remainP, 512,  512);
  wpack_kernel<<<512,  blk, 0, stream>>>(last_w,   lastP,   512,  512);

  // ---- preamble -----------------------------------------------------------
  loc_kernel<<<B_, 512, 0, stream>>>(xyrot, loc1_w, loc1_b, loc2_w, loc2_b, locB);
  pemb_kernel<<<B_ * 14, 512, 0, stream>>>(queue, remains, piece_emb, locB, pembB);
  gemm_bias_wmma<<<dim3(14, 32), blk, 0, stream>>>(pembB, rotP, rot_b, rots,
                                                   B_ * 14, 2048, 512);
  gemm_bias_wmma<<<dim3(56, 2), blk, 0, stream>>>(rots, kP, k_b, kAll,
                                                  B_ * 56, 128, 512);
  gemm_bias_wmma<<<dim3(56, 2), blk, 0, stream>>>(rots, vP, v_b, vAll,
                                                  B_ * 56, 128, 512);
  gconv_kernel<<<B_, 512, 0, stream>>>(garbage, g1_w, g1_b, gconv);
  cnn_a_kernel<<<B_, 256, 0, stream>>>(board, dw1_w, dw1_b, pw1_w, pw1_b,
                                       bn1_g, bn1_b, a1);
  cnn_b_kernel<<<B_ * 16, 256, 0, stream>>>(a1, dw2_w, dw2_b, d2b);
  cnn_state_kernel<<<B_ * 200, 512, 0, stream>>>(d2b, pw2_w, pw2_b, bn2_g, bn2_b,
                                                 gconv, g2_w, g2_b, state);

  // ---- 7-iteration loop (attention collapses to v-row broadcast) ----------
  for (int i = 0; i < 7; ++i) {
    vrow_gather_kernel<<<B_, 512, 0, stream>>>(vAll, i, vrow);
    gemm_bias_wmma<<<dim3(1, 8), blk, 0, stream>>>(vrow, qkvP, qkv_b, nsrow,
                                                   B_, 512, 512);
    ln_res_kernel<<<NROW, 256, 0, stream>>>(state, nsrow, 1, an_g, an_b);
    gemm_bias_wmma<<<dim3(200, 8), blk, 0, stream>>>(state, queueP, queue_b,
                                                     tmpA, NROW, 512, 512);
    ln_res_kernel<<<NROW, 256, 0, stream>>>(state, tmpA, 0, fn_g, fn_b);
  }

  // ---- final attention (Kn = 7) -------------------------------------------
  gemm_bias_wmma<<<dim3(200, 8), blk, 0, stream>>>(state, qP, q_b, tmpA,
                                                   NROW, 512, 512);
  bias_h0_kernel<<<NROW, 256, 0, stream>>>(state, fb_w0, fh0);
  bias_mid_kernel<<<B_, 256, 0, stream>>>(fh0, fb_w1, fb_w2, fb_b2, fb_g1, fb_be1,
                                          fb_w3, fb_b3, fb_g2, fb_be2, fh3);
  bias_w4_kernel<<<5600, 256, 0, stream>>>(fh3, fb_w4, fb_b4, fbias);
  final_attn_kernel<<<NROW, 256, 0, stream>>>(tmpA, kAll, vAll, fbias);
  gemm_bias_wmma<<<dim3(200, 8), blk, 0, stream>>>(tmpA, qkvP, qkv_b, tmpB,
                                                   NROW, 512, 512);
  ln_res_kernel<<<NROW, 256, 0, stream>>>(state, tmpB, 0, fan_g, fan_b);
  gemm_bias_wmma<<<dim3(200, 8), blk, 0, stream>>>(state, remainP, remain_b,
                                                   tmpA, NROW, 512, 512);
  ln_res_kernel<<<NROW, 256, 0, stream>>>(state, tmpA, 0, ffn_g, ffn_b);

  // ---- pool + head --------------------------------------------------------
  pool_kernel<<<B_ * 50, 512, 0, stream>>>(state, pooled);
  gemm_bias_wmma<<<dim3(50, 8), blk, 0, stream>>>(pooled, lastP, last_b, out,
                                                  B_ * 50, 512, 512);
}
